// HLADecayLayer_62302795596551
// MI455X (gfx1250) — matrix operand order repaired and checked by reading, not compiled, not run on hardware
//
#include <hip/hip_runtime.h>

typedef __attribute__((ext_vector_type(16))) __bf16 v16bf;
typedef __attribute__((ext_vector_type(8)))  __bf16 v8bf;
typedef __attribute__((ext_vector_type(8)))  float  v8f;
typedef __attribute__((ext_vector_type(4)))  unsigned int v4u;
typedef __attribute__((ext_vector_type(8)))  int v8i;
typedef __attribute__((ext_vector_type(4)))  int v4i;

#define GAMMA 0.99f

// fp32 -> bf16 with round-to-nearest-even (bit trick)
__device__ __forceinline__ unsigned short f2bf(float x) {
  unsigned int u = __float_as_uint(x);
  u += 0x7FFFu + ((u >> 16) & 1u);
  return (unsigned short)(u >> 16);
}

__device__ __forceinline__ v16bf join16(v8bf lo, v8bf hi) {
  return __builtin_shufflevector(lo, hi, 0, 1, 2, 3, 4, 5, 6, 7,
                                          8, 9, 10, 11, 12, 13, 14, 15);
}

// ---------------------------------------------------------------------------
// Tensor Data Mover: async 2D tile load (rows x 32 bf16) global -> LDS.
// D# packing per CDNA5 ISA (2-D tensor, groups 2/3 zero).
// ---------------------------------------------------------------------------
__device__ __forceinline__ void tdm_load_tile_64x32(unsigned lds_off,
                                                    const unsigned short* gptr,
                                                    unsigned tdim0, unsigned tdim1,
                                                    unsigned stride0) {
  unsigned long long ga = (unsigned long long)(uintptr_t)gptr;
  v4u g0;
  g0[0] = 1u;                                         // count=1, load, user mode
  g0[1] = lds_off;                                    // lds_addr [63:32]
  g0[2] = (unsigned)(ga & 0xFFFFFFFFu);               // global_addr [95:64]
  g0[3] = (unsigned)((ga >> 32) & 0x1FFFFFFu) | (2u << 30);  // addr hi + type=2
  v8i g1;
  g1[0] = (int)(1u << 16);                            // data_size=1 (2 bytes)
  g1[1] = (int)((tdim0 & 0xFFFFu) << 16);             // tensor_dim0[15:0] @63:48
  g1[2] = (int)(((tdim0 >> 16) & 0xFFFFu) | ((tdim1 & 0xFFFFu) << 16));
  g1[3] = (int)(((tdim1 >> 16) & 0xFFFFu) | (32u << 16));    // tile_dim0 = 32
  g1[4] = (int)64u;                                   // tile_dim1 = 64, tile_dim2 = 0
  g1[5] = (int)stride0;                               // tensor_dim0_stride[31:0]
  g1[6] = 0;
  g1[7] = 0;
  v4i z = {0, 0, 0, 0};
#if defined(__clang_major__) && (__clang_major__ >= 23)
  v8i z8 = {0, 0, 0, 0, 0, 0, 0, 0};
  __builtin_amdgcn_tensor_load_to_lds(g0, g1, z, z, z8, 0);
#else
  __builtin_amdgcn_tensor_load_to_lds(g0, g1, z, z, 0);
#endif
}

// ---------------------------------------------------------------------------
// One-time conversions: fp32 -> bf16 copy, and fp32 -> bf16 transpose.
// ---------------------------------------------------------------------------
__global__ __launch_bounds__(256)
void cvt_bf16(const float* __restrict__ in, unsigned short* __restrict__ out) {
  size_t i = (size_t)blockIdx.x * 256 + threadIdx.x;   // one float4 per thread
  float4 xv = *reinterpret_cast<const float4*>(in + i * 4);
  unsigned long long p =
      (unsigned long long)((unsigned)f2bf(xv.x) | ((unsigned)f2bf(xv.y) << 16)) |
      ((unsigned long long)((unsigned)f2bf(xv.z) | ((unsigned)f2bf(xv.w) << 16)) << 32);
  *reinterpret_cast<unsigned long long*>(out + i * 4) = p;
}

__global__ __launch_bounds__(256)
void cvtT_bf16(const float* __restrict__ in, unsigned short* __restrict__ out,
               int R, int C) {   // out[c*R + r] = bf16(in[r*C + c])
  int idx = blockIdx.x * 256 + threadIdx.x;
  int r = idx / C, c = idx % C;
  out[(size_t)c * R + r] = f2bf(in[(size_t)r * C + c]);
}

// ---------------------------------------------------------------------------
// Tiled bf16 WMMA GEMM with TDM-fed, double-buffered LDS tiles.
//   Out[M,N] = A[M,K] * Bt[N,K]^T   (A, Bt bf16 row-major; Out fp32)
// Block = 128 threads (4 waves), tile 64x64, K-step 32.
// MODE==1 scatters output into [B=4][H=16][T=1024][32] layout.
// ---------------------------------------------------------------------------
template <int MODE>
__global__ __launch_bounds__(128)
void gemm_bf16t(const unsigned short* __restrict__ A,
                const unsigned short* __restrict__ Bt,
                float* __restrict__ Out, int M, int N, int K) {
  __shared__ __align__(16) unsigned short Ah[2][64 * 32];
  __shared__ __align__(16) unsigned short Bh[2][64 * 32];

  const int tid  = threadIdx.x;
  const int lane = tid & 31;
  const int wave = tid >> 5;
  const int hi   = lane >> 4;        // WMMA lane-half selector
  const int m    = lane & 15;
  const int wm   = (wave >> 1) * 32; // wave's 32x32 subtile
  const int wn   = (wave & 1) * 32;
  const int m0   = blockIdx.y * 64;
  const int n0   = blockIdx.x * 64;
  const int nk   = K >> 5;

  v8f acc[2][2] = {};

  if (wave == 0) {  // prefetch tile 0 via Tensor Data Mover
    tdm_load_tile_64x32((unsigned)(uintptr_t)&Ah[0][0], A + (size_t)m0 * K, K, M, K);
    tdm_load_tile_64x32((unsigned)(uintptr_t)&Bh[0][0], Bt + (size_t)n0 * K, K, N, K);
  }

  for (int i = 0; i < nk; ++i) {
    if (wave == 0) {
      if (i + 1 < nk) {   // issue next tile pair, then wait for current pair
        int k1 = (i + 1) << 5;
        tdm_load_tile_64x32((unsigned)(uintptr_t)&Ah[(i + 1) & 1][0],
                            A + (size_t)m0 * K + k1, K, M, K);
        tdm_load_tile_64x32((unsigned)(uintptr_t)&Bh[(i + 1) & 1][0],
                            Bt + (size_t)n0 * K + k1, K, N, K);
        __builtin_amdgcn_s_wait_tensorcnt(2);
      } else {
        __builtin_amdgcn_s_wait_tensorcnt(0);
      }
    }
    __syncthreads();   // current buffer ready for all waves

    const unsigned short* Ab = Ah[i & 1];
    const unsigned short* Bb = Bh[i & 1];

    // A fragments (16x32 bf16): lane<16 -> row m, K{0..7,16..23}; lane>=16 -> K{8..15,24..31}
    v16bf fa[2], fb[2];
#pragma unroll
    for (int ti = 0; ti < 2; ++ti) {
      const unsigned short* rp = &Ab[(wm + 16 * ti + m) * 32];
      fa[ti] = join16(*reinterpret_cast<const v8bf*>(rp + 8 * hi),
                      *reinterpret_cast<const v8bf*>(rp + 16 + 8 * hi));
    }
    // B fragments (32x16): lane col = lane%16, K{0..7,16..23} / K{8..15,24..31} by hi (row of Bt)
#pragma unroll
    for (int tj = 0; tj < 2; ++tj) {
      const unsigned short* rp = &Bb[(wn + 16 * tj + m) * 32 + 8 * hi];
      fb[tj] = join16(*reinterpret_cast<const v8bf*>(rp),
                      *reinterpret_cast<const v8bf*>(rp + 16));
    }
#pragma unroll
    for (int ti = 0; ti < 2; ++ti)
#pragma unroll
      for (int tj = 0; tj < 2; ++tj)
        acc[ti][tj] = __builtin_amdgcn_wmma_f32_16x16x32_bf16(
            false, fa[ti], false, fb[tj], (short)0, acc[ti][tj], false, false);
    __syncthreads();   // done reading buffer before TDM overwrites it
  }

  // Epilogue: C/D layout row = r + 8*hi, col = lane%16 within each 16x16 tile.
#pragma unroll
  for (int ti = 0; ti < 2; ++ti)
#pragma unroll
    for (int tj = 0; tj < 2; ++tj)
#pragma unroll
      for (int r = 0; r < 8; ++r) {
        int row = m0 + wm + 16 * ti + r + 8 * hi;
        int col = n0 + wn + 16 * tj + m;
        float val = acc[ti][tj][r];
        if constexpr (MODE == 1) {  // scatter into [b][h][t][d]
          int bb = row >> 10, tt = row & 1023;
          int hh = col >> 5,  dd = col & 31;
          Out[(((size_t)bb * 16 + hh) * 1024 + tt) * 32 + dd] = val;
        } else {
          Out[(size_t)row * N + col] = val;
        }
      }
}

// ---------------------------------------------------------------------------
// Sequential HLA-decay scan: one wave per (b,h) chain (64 blocks x 32 lanes).
// fp32 state in WMMA C/D layout; per step the 32x32x32 product S_C = S_K@C_QV
// runs as 4x v_wmma_f32_16x16x32_bf16 with operands staged through LDS.
// Output written directly as bf16 (A operand of the final GEMM).
// ---------------------------------------------------------------------------
__global__ __launch_bounds__(32)
void hla_scan(const float* __restrict__ qg, const float* __restrict__ kg,
              const float* __restrict__ vg, unsigned short* __restrict__ attn) {
  __shared__ __align__(16) unsigned short SKh[32 * 32]; // row-major bf16
  __shared__ __align__(16) unsigned short Ct [32 * 32]; // Ct[col][row] bf16

  const int bh   = blockIdx.x;
  const int b    = bh >> 4;
  const int h    = bh & 15;
  const int lane = threadIdx.x;
  const int hi   = lane >> 4;
  const int m    = lane & 15;

  const float* qp = qg + (size_t)bh * 1024 * 32;
  const float* kp = kg + (size_t)bh * 1024 * 32;
  const float* vp = vg + (size_t)bh * 1024 * 32;
  unsigned short* op = attn + (size_t)b * 1024 * 512 + h * 32;

  float SK[2][2][8] = {}, Cm[2][2][8] = {}, G[2][2][8] = {};

  for (int t = 0; t < 1024; ++t) {
    float kreg = kp[t * 32 + lane];
    float qreg = qp[t * 32 + lane];
    float vreg = vp[t * 32 + lane];

    // Broadcast row values k[row], q[row] for this lane's rows (row = 16ti+r+8hi)
    float ksel[2][8], qsel[2][8];
#pragma unroll
    for (int ti = 0; ti < 2; ++ti)
#pragma unroll
      for (int r = 0; r < 8; ++r) {
        int src = 16 * ti + r + 8 * hi;
        ksel[ti][r] = __shfl(kreg, src, 32);
        qsel[ti][r] = __shfl(qreg, src, 32);
      }
    float vt[2], kt2[2];
#pragma unroll
    for (int tj = 0; tj < 2; ++tj) {
      vt[tj]  = __shfl(vreg, 16 * tj + m, 32);  // v[col]
      kt2[tj] = __shfl(kreg, 16 * tj + m, 32);  // k[col]
    }

    // k_C[col] = gamma * sum_row k[row] * C_old[row][col]
    float kc[2];
#pragma unroll
    for (int tj = 0; tj < 2; ++tj) {
      float p = 0.f;
#pragma unroll
      for (int ti = 0; ti < 2; ++ti)
#pragma unroll
        for (int r = 0; r < 8; ++r)
          p = fmaf(ksel[ti][r], Cm[ti][tj][r], p);
      p += __shfl_xor(p, 16, 32);
      kc[tj] = GAMMA * p;
    }

    // Decayed rank-1 state updates (fp32)
#pragma unroll
    for (int ti = 0; ti < 2; ++ti)
#pragma unroll
      for (int tj = 0; tj < 2; ++tj)
#pragma unroll
        for (int r = 0; r < 8; ++r) {
          Cm[ti][tj][r] = fmaf(GAMMA, Cm[ti][tj][r], qsel[ti][r] * vt[tj]);
          G [ti][tj][r] = fmaf(GAMMA, G [ti][tj][r], ksel[ti][r] * kc[tj]);
          SK[ti][tj][r] = fmaf(GAMMA, SK[ti][tj][r], ksel[ti][r] * kt2[tj]);
        }

    // Stage S_K (A operand, row-major) and C_QV (B operand, transposed) as bf16
#pragma unroll
    for (int ti = 0; ti < 2; ++ti)
#pragma unroll
      for (int tj = 0; tj < 2; ++tj)
#pragma unroll
        for (int r = 0; r < 8; ++r) {
          int row = 16 * ti + r + 8 * hi;
          int col = 16 * tj + m;
          SKh[row * 32 + col] = f2bf(SK[ti][tj][r]);
          Ct [col * 32 + row] = f2bf(Cm[ti][tj][r]);
        }
    __syncthreads();

    v16bf fa[2], fb[2];
#pragma unroll
    for (int ti = 0; ti < 2; ++ti) {
      const unsigned short* rp = &SKh[(16 * ti + m) * 32];
      fa[ti] = join16(*reinterpret_cast<const v8bf*>(rp + 8 * hi),
                      *reinterpret_cast<const v8bf*>(rp + 16 + 8 * hi));
    }
#pragma unroll
    for (int tj = 0; tj < 2; ++tj) {
      const unsigned short* rp = &Ct[(16 * tj + m) * 32 + 8 * hi];
      fb[tj] = join16(*reinterpret_cast<const v8bf*>(rp),
                      *reinterpret_cast<const v8bf*>(rp + 16));
    }

    v8f d[2][2];
#pragma unroll
    for (int ti = 0; ti < 2; ++ti)
#pragma unroll
      for (int tj = 0; tj < 2; ++tj) {
        v8f z = {};
        d[ti][tj] = __builtin_amdgcn_wmma_f32_16x16x32_bf16(
            false, fa[ti], false, fb[tj], (short)0, z, false, false);
      }
    __syncthreads();   // buffers rewritten next iteration

    // o[col] = sum_row q[row] * (S_C - G)[row][col]
    float o2[2];
#pragma unroll
    for (int tj = 0; tj < 2; ++tj) {
      float p = 0.f;
#pragma unroll
      for (int ti = 0; ti < 2; ++ti)
#pragma unroll
        for (int r = 0; r < 8; ++r)
          p = fmaf(qsel[ti][r], d[ti][tj][r] - G[ti][tj][r], p);
      p += __shfl_xor(p, 16, 32);
      o2[tj] = p;
    }
    op[(size_t)t * 512 + lane] = f2bf(hi ? o2[1] : o2[0]);  // col == lane
  }
}

// ---------------------------------------------------------------------------
extern "C" void kernel_launch(void* const* d_in, const int* in_sizes, int n_in,
                              void* d_out, int out_size, void* d_ws, size_t ws_size,
                              hipStream_t stream) {
  const float* x  = (const float*)d_in[0];
  const float* Wq = (const float*)d_in[1];
  const float* Wk = (const float*)d_in[2];
  const float* Wv = (const float*)d_in[3];
  const float* Wo = (const float*)d_in[4];
  float* out = (float*)d_out;

  // Workspace carve-up (bytes)
  char* p = (char*)d_ws;
  float* q_ws = (float*)p;              p += (size_t)8 << 20;  // [b][h][t][32] fp32
  float* k_ws = (float*)p;              p += (size_t)8 << 20;
  float* v_ws = (float*)p;              p += (size_t)8 << 20;
  unsigned short* attn_h = (unsigned short*)p; p += (size_t)4 << 20; // [4096][512] bf16
  unsigned short* xh  = (unsigned short*)p;    p += (size_t)8 << 20; // [4096][1024] bf16
  unsigned short* WqT = (unsigned short*)p;    p += (size_t)1 << 20; // [512][1024] bf16
  unsigned short* WkT = (unsigned short*)p;    p += (size_t)1 << 20;
  unsigned short* WvT = (unsigned short*)p;    p += (size_t)1 << 20;
  unsigned short* WoT = (unsigned short*)p;    p += (size_t)1 << 20; // [1024][512] bf16

  // One-time bf16 conversion / weight transposition (makes every GEMM tile a
  // row-major 2D bf16 tile the Tensor Data Mover can DMA directly into LDS).
  cvt_bf16<<<dim3(4096), dim3(256), 0, stream>>>(x, xh);                 // 4M elems
  cvtT_bf16<<<dim3(2048), dim3(256), 0, stream>>>(Wq, WqT, 1024, 512);
  cvtT_bf16<<<dim3(2048), dim3(256), 0, stream>>>(Wk, WkT, 1024, 512);
  cvtT_bf16<<<dim3(2048), dim3(256), 0, stream>>>(Wv, WvT, 1024, 512);
  cvtT_bf16<<<dim3(2048), dim3(256), 0, stream>>>(Wo, WoT, 512, 1024);

  // QKV projections: [4096x1024] x [1024x512], scattered to [b][h][t][d]
  gemm_bf16t<1><<<dim3(8, 64), dim3(128), 0, stream>>>(xh, WqT, q_ws, 4096, 512, 1024);
  gemm_bf16t<1><<<dim3(8, 64), dim3(128), 0, stream>>>(xh, WkT, k_ws, 4096, 512, 1024);
  gemm_bf16t<1><<<dim3(8, 64), dim3(128), 0, stream>>>(xh, WvT, v_ws, 4096, 512, 1024);

  // Sequential decay scan: one wave per (b,h) chain
  hla_scan<<<dim3(64), dim3(32), 0, stream>>>(q_ws, k_ws, v_ws, attn_h);

  // Output projection: [4096x512] x [512x1024] -> d_out (fp32)
  gemm_bf16t<0><<<dim3(16, 64), dim3(128), 0, stream>>>(attn_h, WoT, out, 4096, 1024, 512);
}
